// NoSpikingLIF_32701880992464
// MI455X (gfx1250) — matrix operand order repaired and checked by reading, not compiled, not run on hardware
//
#include <hip/hip_runtime.h>

// ---------------- problem constants (from reference) ----------------
#define L_T   1024
#define N_B   64
#define IN_F  1024
#define OUT_F 1024
#define LN    (L_T * N_B)          // 65536 GEMM rows

// ---------------- GEMM tiling ----------------
#define BM 128
#define BN 128
#define BK 32
#define LDS_STRIDE 40              // halves per row (32 + 8 pad) -> conflict-free gathers

// ---------------- vector types (trivial ext-vectors; union-safe) ----------------
typedef __attribute__((ext_vector_type(16))) __bf16       v16bf;
typedef __attribute__((ext_vector_type(8)))  float        v8f;
typedef __attribute__((ext_vector_type(4)))  float        v4f;
typedef __attribute__((ext_vector_type(4)))  unsigned int v4u;
typedef __attribute__((ext_vector_type(2)))  unsigned int v2u;

union FragBF { v4u q[2]; v16bf v; };   // 32 bytes = 8 VGPRs = one 16x32 bf16 fragment

// ---------------- fp32 <-> bf16 split helpers (bit-exact, RNE) ----------------
__device__ __forceinline__ unsigned short f32_to_bf16_rne(float f) {
  unsigned u = __float_as_uint(f);
  unsigned r = 0x7FFFu + ((u >> 16) & 1u);
  return (unsigned short)((u + r) >> 16);
}
__device__ __forceinline__ float bf16u_to_f32(unsigned short h) {
  return __uint_as_float(((unsigned)h) << 16);
}

// split a float4 into bf16 hi/lo and store 8B packed
__device__ __forceinline__ void split4_store(v4f v, unsigned short* hi, unsigned short* lo) {
  float f0 = v.x, f1 = v.y, f2 = v.z, f3 = v.w;
  unsigned short h0 = f32_to_bf16_rne(f0), h1 = f32_to_bf16_rne(f1);
  unsigned short h2 = f32_to_bf16_rne(f2), h3 = f32_to_bf16_rne(f3);
  unsigned short l0 = f32_to_bf16_rne(f0 - bf16u_to_f32(h0));
  unsigned short l1 = f32_to_bf16_rne(f1 - bf16u_to_f32(h1));
  unsigned short l2 = f32_to_bf16_rne(f2 - bf16u_to_f32(h2));
  unsigned short l3 = f32_to_bf16_rne(f3 - bf16u_to_f32(h3));
  v2u hp = { (unsigned)h0 | ((unsigned)h1 << 16), (unsigned)h2 | ((unsigned)h3 << 16) };
  v2u lp = { (unsigned)l0 | ((unsigned)l1 << 16), (unsigned)l2 | ((unsigned)l3 << 16) };
  *(v2u*)hi = hp;
  *(v2u*)lo = lp;
}

// ======================================================================
// Kernel 0 (prologue): split W (fp32, 4MB) once into bf16 hi/lo in d_ws.
// Removes ~512M redundant per-block conversions from the GEMM hot loop.
// ======================================================================
__global__ __launch_bounds__(256) void split_W(
    const float* __restrict__ W,          // [OUT_F][IN_F]
    unsigned short* __restrict__ Whi,     // [OUT_F][IN_F] bf16 bits
    unsigned short* __restrict__ Wlo)     // [OUT_F][IN_F] bf16 bits
{
  const size_t i4 = ((size_t)blockIdx.x * blockDim.x + threadIdx.x) * 4;
  v4f v = *(const v4f*)(W + i4);
  split4_store(v, Whi + i4, Wlo + i4);
}

// ======================================================================
// Kernel 1: proj[m][o] = sum_i spk[m][i] * W[o][i]   (fp32 via bf16x3)
// m = l*N + n flattened.  proj is written straight into d_out (out_seq slot).
// A: fp32 -> split in-flight -> LDS.  B: pre-split bf16 copied global->LDS.
// ======================================================================
__global__ __launch_bounds__(256) void gemm_split_bf16(
    const float* __restrict__ spk,            // [LN][IN_F] fp32
    const unsigned short* __restrict__ Whi,   // [OUT_F][IN_F] bf16
    const unsigned short* __restrict__ Wlo,   // [OUT_F][IN_F] bf16
    float* __restrict__ proj)                 // [LN][OUT_F]  (== d_out)
{
  __shared__ unsigned short shAhi[BM * LDS_STRIDE];
  __shared__ unsigned short shAlo[BM * LDS_STRIDE];
  __shared__ unsigned short shBhi[BN * LDS_STRIDE];
  __shared__ unsigned short shBlo[BN * LDS_STRIDE];

  const int tid  = threadIdx.x;
  const int lane = tid & 31;
  const int wid  = tid >> 5;       // 0..7
  const int wm   = wid & 1;        // 2 row-slabs of 64
  const int wn   = wid >> 1;       // 4 col-slabs of 32

  // nblk fastest in dispatch order: the 8 blocks sharing one 512KB A-tile are
  // co-resident, so spk reuse is served out of the 192MB L2 (HBM reads ~1x).
  const int mblk = (int)blockIdx.x / (OUT_F / BN);
  const int nblk = (int)blockIdx.x % (OUT_F / BN);
  const int m0 = mblk * BM;
  const int o0 = nblk * BN;

  // cooperative fill mapping: thread -> (row 0..127, 16-element half-row)
  const int frow = tid >> 1;
  const int fk   = (tid & 1) * 16;

  // per-lane fragment gather coordinates (ISA 7.12.2 layouts)
  const int arow = lane & 15;            // A: M row
  const int akb  = (lane >> 4) * 8;      // A: K runs [akb, akb+8) and [akb+16, akb+24)
  const int bcol = lane & 15;            // B: N col
  const int bkb  = (lane >> 4) * 8;      // B: K runs [bkb, bkb+8) and [bkb+16, bkb+24)

  const v8f vzero = {0.f, 0.f, 0.f, 0.f, 0.f, 0.f, 0.f, 0.f};
  v8f acc[4][2];
#pragma unroll
  for (int i = 0; i < 4; ++i)
#pragma unroll
    for (int j = 0; j < 2; ++j) acc[i][j] = vzero;

  for (int k0 = 0; k0 < IN_F; k0 += BK) {
    const float*          aSrc  = spk + (size_t)(m0 + frow) * IN_F + k0 + fk;
    const unsigned short* bhSrc = Whi + (size_t)(o0 + frow) * IN_F + k0 + fk;
    const unsigned short* blSrc = Wlo + (size_t)(o0 + frow) * IN_F + k0 + fk;

    if (k0 + BK < IN_F) {                       // pull next A K-tile toward L2/L0
      __builtin_prefetch(aSrc + BK, 0, 3);      // -> global_prefetch_b8
    }

    // ---- A: fp32 -> bf16 hi/lo split into LDS ----
#pragma unroll
    for (int j = 0; j < 4; ++j) {
      v4f av = *(const v4f*)(aSrc + 4 * j);
      split4_store(av, &shAhi[frow * LDS_STRIDE + fk + 4 * j],
                       &shAlo[frow * LDS_STRIDE + fk + 4 * j]);
    }
    // ---- B: pre-split bf16, straight copy to LDS (no math) ----
    *(v4u*)&shBhi[frow * LDS_STRIDE + fk]     = *(const v4u*)(bhSrc);
    *(v4u*)&shBhi[frow * LDS_STRIDE + fk + 8] = *(const v4u*)(bhSrc + 8);
    *(v4u*)&shBlo[frow * LDS_STRIDE + fk]     = *(const v4u*)(blSrc);
    *(v4u*)&shBlo[frow * LDS_STRIDE + fk + 8] = *(const v4u*)(blSrc + 8);
    __syncthreads();

    // A fragments for this wave's 64 rows (hi and lo)
    FragBF ahi[4], alo[4];
#pragma unroll
    for (int mi = 0; mi < 4; ++mi) {
      const int r = wm * 64 + mi * 16 + arow;
      ahi[mi].q[0] = *(const v4u*)&shAhi[r * LDS_STRIDE + akb];
      ahi[mi].q[1] = *(const v4u*)&shAhi[r * LDS_STRIDE + akb + 16];
      alo[mi].q[0] = *(const v4u*)&shAlo[r * LDS_STRIDE + akb];
      alo[mi].q[1] = *(const v4u*)&shAlo[r * LDS_STRIDE + akb + 16];
    }

#pragma unroll
    for (int ni = 0; ni < 2; ++ni) {
      const int c = wn * 32 + ni * 16 + bcol;
      FragBF bhi, blo;
      bhi.q[0] = *(const v4u*)&shBhi[c * LDS_STRIDE + bkb];
      bhi.q[1] = *(const v4u*)&shBhi[c * LDS_STRIDE + bkb + 16];
      blo.q[0] = *(const v4u*)&shBlo[c * LDS_STRIDE + bkb];
      blo.q[1] = *(const v4u*)&shBlo[c * LDS_STRIDE + bkb + 16];
#pragma unroll
      for (int mi = 0; mi < 4; ++mi) {
        // fp32-accurate product: Ahi*Bhi + Ahi*Blo + Alo*Bhi
        acc[mi][ni] = __builtin_amdgcn_wmma_f32_16x16x32_bf16(
            false, ahi[mi].v, false, bhi.v, (short)0, acc[mi][ni], false, false);
        acc[mi][ni] = __builtin_amdgcn_wmma_f32_16x16x32_bf16(
            false, ahi[mi].v, false, blo.v, (short)0, acc[mi][ni], false, false);
        acc[mi][ni] = __builtin_amdgcn_wmma_f32_16x16x32_bf16(
            false, alo[mi].v, false, bhi.v, (short)0, acc[mi][ni], false, false);
      }
    }
    __syncthreads();
  }

  // C layout: lanes 0-15: VGPR r -> M=r ; lanes 16-31: VGPR r -> M=8+r ; N = lane%16
  const int mofs = (lane >> 4) * 8;
#pragma unroll
  for (int mi = 0; mi < 4; ++mi) {
#pragma unroll
    for (int ni = 0; ni < 2; ++ni) {
      const int mbase = m0 + wm * 64 + mi * 16 + mofs;
      const int ob    = o0 + wn * 32 + ni * 16 + (lane & 15);
#pragma unroll
      for (int r = 0; r < 8; ++r) {
        proj[(size_t)(mbase + r) * OUT_F + ob] = acc[mi][ni][r];
      }
    }
  }
}

// ======================================================================
// Kernel 2: in-place LIF scan over L.  One thread per (n,o) chain;
// wave-contiguous in o -> 128B coalesced per access.  Bias folded here.
//   m = beta*m + (proj + b);  out[t] = m;  final_mem appended at tail.
// ======================================================================
__global__ __launch_bounds__(256) void lif_scan(
    float* __restrict__ out,          // d_out: [L][N][OUT] proj -> out_seq, then [N][OUT] final
    const float* __restrict__ mem0,   // [N][OUT]
    const float* __restrict__ b,      // [OUT]
    const float* __restrict__ beta)   // [OUT]
{
  const int tid = (int)blockIdx.x * (int)blockDim.x + (int)threadIdx.x; // 0..N*OUT-1
  const int o   = tid & (OUT_F - 1);
  float m  = mem0[tid];
  const float bb = b[o];
  const float bt = beta[o];
  float* p = out + tid;
#pragma unroll 4
  for (int t = 0; t < L_T; ++t) {
    const size_t idx = (size_t)t * (N_B * OUT_F);
    float cur = p[idx] + bb;
    m = fmaf(bt, m, cur);
    p[idx] = m;
  }
  out[(size_t)L_T * N_B * OUT_F + tid] = m;
}

// ======================================================================
extern "C" void kernel_launch(void* const* d_in, const int* in_sizes, int n_in,
                              void* d_out, int out_size, void* d_ws, size_t ws_size,
                              hipStream_t stream) {
  (void)in_sizes; (void)n_in; (void)out_size; (void)ws_size;
  const float* spk  = (const float*)d_in[0];   // [L][N][IN]
  const float* mem  = (const float*)d_in[1];   // [N][OUT]
  const float* W    = (const float*)d_in[2];   // [OUT][IN]
  const float* b    = (const float*)d_in[3];   // [OUT]
  const float* beta = (const float*)d_in[4];   // [OUT]
  float* out = (float*)d_out;

  // 4 MiB of scratch: bf16 hi/lo split of W (L2-resident for the whole GEMM)
  unsigned short* Whi = (unsigned short*)d_ws;
  unsigned short* Wlo = Whi + (size_t)OUT_F * IN_F;

  dim3 grid0(((size_t)OUT_F * IN_F) / (256 * 4));  // 1024 blocks
  split_W<<<grid0, 256, 0, stream>>>(W, Whi, Wlo);

  dim3 grid1((LN / BM) * (OUT_F / BN));        // 512 * 8 = 4096 blocks
  gemm_split_bf16<<<grid1, 256, 0, stream>>>(spk, Whi, Wlo, out);

  dim3 grid2((N_B * OUT_F) / 256);             // 256 blocks, 65536 chains
  lif_scan<<<grid2, 256, 0, stream>>>(out, mem, b, beta);
}
